// S2IPLLM_89472758710943
// MI455X (gfx1250) — matrix operand order repaired and checked by reading, not compiled, not run on hardware
//
#include <hip/hip_runtime.h>
#include <hip/hip_bf16.h>

// ---------------------------------------------------------------------------
// Problem constants
// ---------------------------------------------------------------------------
#define BATCH   64
#define NVARS   7
#define NSEQ    (BATCH * NVARS)   // 448
#define SEQLEN  512
#define PATCH   16
#define STRIDE  8
#define PNUM    64                // (512-16)/8 + 2
#define DMODEL  768
#define NHEAD   12
#define HDIM    64
#define TSEQ    68                // TOP_K + PNUM
#define NLAYERS 6
#define FDIM    3072
#define POOL    1000
#define TOPK    4
#define PREDLEN 96
#define TRENDW  24
#define SEASP   24
#define NROWS   (NSEQ * TSEQ)     // 30464
#define KOUT    ((DMODEL / 3) * (PNUM + TOPK))  // 17408

typedef __attribute__((ext_vector_type(16))) __bf16 v16bf;
typedef __attribute__((ext_vector_type(8)))  __bf16 v8bf;
typedef __attribute__((ext_vector_type(8)))  float  v8f;

__device__ __forceinline__ float gelu_tanh(float x) {
    float x3 = x * x * x;
    return 0.5f * x * (1.0f + tanhf(0.7978845608028654f * (x + 0.044715f * x3)));
}

// 16-byte global->LDS async copy (Tensor/async path, ASYNCcnt-tracked).
__device__ __forceinline__ void async_copy16(const __bf16* g, const __bf16* l) {
    unsigned lds_addr = (unsigned)(unsigned long long)(uintptr_t)l;  // low 32b of flat = LDS offset
    unsigned long long ga = (unsigned long long)(uintptr_t)g;
    asm volatile("global_load_async_to_lds_b128 %0, %1, off"
                 :: "v"(lds_addr), "v"(ga) : "memory");
}

// ---------------------------------------------------------------------------
// bf16 WMMA GEMM with async-LDS double-buffered staging.
//   A : M x K row-major bf16 (K multiple of 32; rows up to gridY*128 readable)
//   Bt: N x K row-major bf16 (transposed weights; N padded to 128 multiple)
//   C : fp32 or bf16 (obf), + bias, optional tanh-GELU, optional fp32 residual
// Block: 256 threads = 8 waves; tile 128(M) x 128(N), K-step 32.
// Wave (4x2) owns 32x64 = 2x4 v_wmma_f32_16x16x32_bf16.
// ---------------------------------------------------------------------------
#define BM 128
#define BN 128
#define BK 32

__global__ __launch_bounds__(256) void gemm_bf16_wmma(
    const __bf16* __restrict__ A, const __bf16* __restrict__ Bt,
    const float* __restrict__ bias, const float* __restrict__ res,
    void* __restrict__ Cv, int M, int N, int K, int act, int obf)
{
    __shared__ alignas(64) __bf16 As[2][BM][BK];
    __shared__ alignas(64) __bf16 Bs[2][BN][BK];

    const int tid   = threadIdx.x;
    const int lane  = tid & 31;
    const int wave  = tid >> 5;
    const int waveM = wave & 3;   // 0..3 (M)
    const int waveN = wave >> 2;  // 0..1 (N)
    const int cM = blockIdx.y * BM;
    const int cN = blockIdx.x * BN;

    // Staging: 16B segments. A tile = 128 rows x 4 segs = 512 segs; 2/thread.
    // Segments (2*tid, 2*tid+1) are row-adjacent (seg index even).
    const int ar = (tid * 2) >> 2;        // row 0..127
    const int as = ((tid * 2) & 3) * 8;   // k offset 0 or 16 (pairs of segs)
    const __bf16* Agp = A  + (long)(cM + ar) * K + as;
    const __bf16* Bgp = Bt + (long)(cN + ar) * K + as;

    v8f acc[2][4] = {};

    const int nk = K / BK;
    // Prologue: issue tile 0 into buffer 0.
    {
        async_copy16(Agp,     &As[0][ar][as]);
        async_copy16(Agp + 8, &As[0][ar][as + 8]);
        async_copy16(Bgp,     &Bs[0][ar][as]);
        async_copy16(Bgp + 8, &Bs[0][ar][as + 8]);
    }

    for (int t = 0; t < nk; t++) {
        const int buf = t & 1;
        if (t + 1 < nk) {
            const long ko = (long)(t + 1) * BK;
            async_copy16(Agp + ko,     &As[buf ^ 1][ar][as]);
            async_copy16(Agp + ko + 8, &As[buf ^ 1][ar][as + 8]);
            async_copy16(Bgp + ko,     &Bs[buf ^ 1][ar][as]);
            async_copy16(Bgp + ko + 8, &Bs[buf ^ 1][ar][as + 8]);
            asm volatile("s_wait_asynccnt 0x4" ::: "memory");  // tile t done (in-order)
        } else {
            asm volatile("s_wait_asynccnt 0x0" ::: "memory");
        }
        __syncthreads();   // tile t visible to all waves

        // Fragments: contiguous k order (k = 16*(lane>>4) + s), same for A and B.
        v16bf afrag[2], bfrag[4];
        #pragma unroll
        for (int mi = 0; mi < 2; mi++) {
            int row = waveM * 32 + mi * 16 + (lane & 15);
            int kb  = (lane >> 4) * 16;                      // lanes16-31: K=16..31
            afrag[mi] = *(const v16bf*)&As[buf][row][kb];
        }
        #pragma unroll
        for (int ni = 0; ni < 4; ni++) {
            int col = waveN * 64 + ni * 16 + (lane & 15);
            int kb  = (lane >> 4) * 16;                      // lanes16-31: K=16..31
            bfrag[ni] = *(const v16bf*)&Bs[buf][col][kb];
        }
        #pragma unroll
        for (int mi = 0; mi < 2; mi++)
            #pragma unroll
            for (int ni = 0; ni < 4; ni++)
                acc[mi][ni] = __builtin_amdgcn_wmma_f32_16x16x32_bf16(
                    false, afrag[mi], false, bfrag[ni],
                    (short)0, acc[mi][ni], false, false);

        __syncthreads();   // protect buf from t+2's async writes
    }

    // Epilogue. C/D layout: VGPR r -> M = (lane>=16?8:0)+r, N = lane&15.
    float*  Cf = (float*)Cv;
    __bf16* Cb = (__bf16*)Cv;
    #pragma unroll
    for (int mi = 0; mi < 2; mi++) {
        #pragma unroll
        for (int ni = 0; ni < 4; ni++) {
            int n  = cN + waveN * 64 + ni * 16 + (lane & 15);
            int mb = cM + waveM * 32 + mi * 16 + ((lane >> 4) * 8);
            float bv = (bias && n < N) ? bias[n] : 0.0f;
            #pragma unroll
            for (int r = 0; r < 8; r++) {
                int m = mb + r;
                if (m < M && n < N) {
                    float v = acc[mi][ni][r] + bv;
                    if (act) v = gelu_tanh(v);
                    if (res) v += res[(long)m * N + n];
                    if (obf) Cb[(long)m * N + n] = (__bf16)v;
                    else     Cf[(long)m * N + n] = v;
                }
            }
        }
    }
}

// ---------------------------------------------------------------------------
// Weight convert + transpose: Wt[n][k] = bf16(W[k][n]); zero-padded.
// ---------------------------------------------------------------------------
__global__ void wtrans_kernel(const float* __restrict__ W, __bf16* __restrict__ Wt,
                              int K, int N, int Kpad, long total)
{
    long idx = (long)blockIdx.x * 256 + threadIdx.x;
    if (idx >= total) return;
    int  k = (int)(idx % Kpad);
    long n = idx / Kpad;
    float v = (k < K && n < N) ? W[(long)k * N + n] : 0.0f;
    Wt[idx] = (__bf16)v;
}

// ---------------------------------------------------------------------------
// Per-(b,m) normalization
// ---------------------------------------------------------------------------
__global__ __launch_bounds__(256) void normalize_kernel(
    const float* __restrict__ x, float* __restrict__ xn,
    float* __restrict__ means, float* __restrict__ stdev)
{
    __shared__ float r1[256], r2[256];
    int n = blockIdx.x, b = n / NVARS, m = n % NVARS;
    float s = 0.f, s2 = 0.f;
    for (int l = threadIdx.x; l < SEQLEN; l += 256) {
        float v = x[((long)b * SEQLEN + l) * NVARS + m];
        s += v; s2 += v * v;
    }
    r1[threadIdx.x] = s; r2[threadIdx.x] = s2;
    __syncthreads();
    for (int st = 128; st > 0; st >>= 1) {
        if (threadIdx.x < st) { r1[threadIdx.x] += r1[threadIdx.x + st]; r2[threadIdx.x] += r2[threadIdx.x + st]; }
        __syncthreads();
    }
    float mu  = r1[0] / SEQLEN;
    float var = r2[0] / SEQLEN - mu * mu;
    float sd  = sqrtf(var + 1e-5f);
    if (threadIdx.x == 0) { means[n] = mu; stdev[n] = sd; }
    for (int l = threadIdx.x; l < SEQLEN; l += 256)
        xn[(long)n * SEQLEN + l] = (x[((long)b * SEQLEN + l) * NVARS + m] - mu) / sd;
}

// ---------------------------------------------------------------------------
// Decomposition: trend / seasonal / residual. dec laid out (NSEQ, 3, L).
// ---------------------------------------------------------------------------
__global__ __launch_bounds__(256) void decompose_kernel(
    const float* __restrict__ xn, float* __restrict__ dec)
{
    __shared__ float xs[SEQLEN];
    __shared__ float tr[SEQLEN];
    __shared__ float gm[SEASP];
    int n = blockIdx.x;
    for (int l = threadIdx.x; l < SEQLEN; l += 256) xs[l] = xn[(long)n * SEQLEN + l];
    __syncthreads();
    for (int l = threadIdx.x; l < SEQLEN; l += 256) {
        int i = l - 12; if (i < 0) i = 0; if (i > SEQLEN - TRENDW) i = SEQLEN - TRENDW;
        float s = 0.f;
        #pragma unroll
        for (int j = 0; j < TRENDW; j++) s += xs[i + j];
        tr[l] = s / (float)TRENDW;
    }
    __syncthreads();
    if (threadIdx.x < SEASP) {
        int s0 = threadIdx.x;
        float acc = 0.f; int cnt = 0;
        for (int l = s0; l < SEQLEN; l += SEASP) { acc += xs[l] - tr[l]; cnt++; }
        gm[s0] = acc / (float)cnt;
    }
    __syncthreads();
    for (int l = threadIdx.x; l < SEQLEN; l += 256) {
        float t = tr[l], se = gm[l % SEASP];
        dec[((long)n * 3 + 0) * SEQLEN + l] = t;
        dec[((long)n * 3 + 1) * SEQLEN + l] = se;
        dec[((long)n * 3 + 2) * SEQLEN + l] = xs[l] - t - se;
    }
}

// ---------------------------------------------------------------------------
// Token matrix (bf16, K padded 48->64): tok[row=n*64+p][c*16+j] ; cols 48..63 = 0
// ---------------------------------------------------------------------------
__global__ void tok_kernel(const float* __restrict__ dec, __bf16* __restrict__ tok)
{
    long idx = (long)blockIdx.x * 256 + threadIdx.x;
    const long total = (long)NSEQ * PNUM * 64;
    if (idx >= total) return;
    int  col = (int)(idx % 64);
    long row = idx / 64;
    int  p = (int)(row % PNUM);
    long n = row / PNUM;
    float v = 0.0f;
    if (col < 48) {
        int c = col / PATCH, j = col % PATCH;
        int l = p * STRIDE + j; if (l > SEQLEN - 1) l = SEQLEN - 1;
        v = dec[(n * 3 + c) * SEQLEN + l];
    }
    tok[idx] = (__bf16)v;
}

// ---------------------------------------------------------------------------
// L2-normalize rows (prompt keys)
// ---------------------------------------------------------------------------
__global__ __launch_bounds__(256) void l2norm_rows(
    const float* __restrict__ in, float* __restrict__ out, int D)
{
    __shared__ float r[256];
    long row = blockIdx.x;
    float s = 0.f;
    for (int i = threadIdx.x; i < D; i += 256) { float v = in[row * D + i]; s += v * v; }
    r[threadIdx.x] = s;
    __syncthreads();
    for (int st = 128; st > 0; st >>= 1) {
        if (threadIdx.x < st) r[threadIdx.x] += r[threadIdx.x + st];
        __syncthreads();
    }
    float inv = rsqrtf(fmaxf(r[0], 1e-12f));
    for (int i = threadIdx.x; i < D; i += 256) out[row * D + i] = in[row * D + i] * inv;
}

// ---------------------------------------------------------------------------
// xq = l2norm(mean over patches of emb)
// ---------------------------------------------------------------------------
__global__ __launch_bounds__(256) void xq_kernel(
    const float* __restrict__ emb, float* __restrict__ xq)
{
    __shared__ float r[256];
    int n = blockIdx.x;
    float ss = 0.f;
    for (int d = threadIdx.x; d < DMODEL; d += 256) {
        float acc = 0.f;
        for (int p = 0; p < PNUM; p++) acc += emb[((long)n * PNUM + p) * DMODEL + d];
        acc *= (1.0f / PNUM);
        xq[(long)n * DMODEL + d] = acc;
        ss += acc * acc;
    }
    r[threadIdx.x] = ss;
    __syncthreads();
    for (int st = 128; st > 0; st >>= 1) {
        if (threadIdx.x < st) r[threadIdx.x] += r[threadIdx.x + st];
        __syncthreads();
    }
    float inv = rsqrtf(fmaxf(r[0], 1e-12f));
    for (int d = threadIdx.x; d < DMODEL; d += 256) xq[(long)n * DMODEL + d] *= inv;
}

__global__ void init_scalar(float* p) { if (blockIdx.x == 0 && threadIdx.x == 0) p[0] = 0.f; }

// ---------------------------------------------------------------------------
// Similarity + top-4 + reduce_sim accumulation
// ---------------------------------------------------------------------------
__global__ __launch_bounds__(256) void topk_kernel(
    const float* __restrict__ xq, const float* __restrict__ keyn,
    int* __restrict__ idx, float* __restrict__ reduce_sim)
{
    __shared__ float sim[POOL];
    __shared__ float rv[256];
    __shared__ int   ri[256];
    int n = blockIdx.x;
    const float* xr = &xq[(long)n * DMODEL];
    for (int j = threadIdx.x; j < POOL; j += 256) {
        const float* kr = &keyn[(long)j * DMODEL];
        float acc = 0.f;
        for (int d = 0; d < DMODEL; d++) acc += xr[d] * kr[d];
        sim[j] = acc;
    }
    __syncthreads();
    float sum0 = 0.f;
    for (int k = 0; k < TOPK; k++) {
        float bv = -1e30f; int bi = 0;
        for (int j = threadIdx.x; j < POOL; j += 256)
            if (sim[j] > bv) { bv = sim[j]; bi = j; }
        rv[threadIdx.x] = bv; ri[threadIdx.x] = bi;
        __syncthreads();
        for (int st = 128; st > 0; st >>= 1) {
            if (threadIdx.x < st && rv[threadIdx.x + st] > rv[threadIdx.x]) {
                rv[threadIdx.x] = rv[threadIdx.x + st]; ri[threadIdx.x] = ri[threadIdx.x + st];
            }
            __syncthreads();
        }
        int best = ri[0];
        if (threadIdx.x == 0) { idx[n * TOPK + k] = best; sum0 += rv[0]; }
        __syncthreads();
        if (threadIdx.x == 0) sim[best] = -1e30f;
        __syncthreads();
    }
    if (threadIdx.x == 0) atomicAdd(reduce_sim, sum0 / (float)NSEQ);
}

// ---------------------------------------------------------------------------
// prompted = concat(bp, emb) + wpe  ->  h (fp32, NSEQ x TSEQ x D)
// ---------------------------------------------------------------------------
__global__ __launch_bounds__(256) void concat_kernel(
    const float* __restrict__ keyn, const int* __restrict__ idx,
    const float* __restrict__ emb, const float* __restrict__ wpe,
    float* __restrict__ h)
{
    int n = blockIdx.x;
    for (int i = threadIdx.x; i < TSEQ * DMODEL; i += 256) {
        int t = i / DMODEL, d = i % DMODEL;
        float v;
        if (t < TOPK) v = keyn[(long)idx[n * TOPK + t] * DMODEL + d];
        else          v = emb[((long)n * PNUM + (t - TOPK)) * DMODEL + d];
        h[((long)n * TSEQ + t) * DMODEL + d] = v + wpe[(long)t * DMODEL + d];
    }
}

// ---------------------------------------------------------------------------
// LayerNorm over last dim (fp32 in -> bf16 out)
// ---------------------------------------------------------------------------
__global__ __launch_bounds__(256) void layernorm_kernel(
    const float* __restrict__ x, const float* __restrict__ g,
    const float* __restrict__ b, __bf16* __restrict__ y, int D)
{
    __shared__ float r1[256], r2[256];
    long row = blockIdx.x;
    const float* xr = x + row * D;
    float s = 0.f, s2 = 0.f;
    for (int i = threadIdx.x; i < D; i += 256) { float v = xr[i]; s += v; s2 += v * v; }
    r1[threadIdx.x] = s; r2[threadIdx.x] = s2;
    __syncthreads();
    for (int st = 128; st > 0; st >>= 1) {
        if (threadIdx.x < st) { r1[threadIdx.x] += r1[threadIdx.x + st]; r2[threadIdx.x] += r2[threadIdx.x + st]; }
        __syncthreads();
    }
    float mu  = r1[0] / D;
    float var = r2[0] / D - mu * mu;
    float inv = rsqrtf(var + 1e-5f);
    for (int i = threadIdx.x; i < D; i += 256)
        y[row * D + i] = (__bf16)((xr[i] - mu) * inv * g[i] + b[i]);
}

// ---------------------------------------------------------------------------
// Fused causal attention per (sequence, head); bf16 qkv in, bf16 out.
// ---------------------------------------------------------------------------
__global__ __launch_bounds__(256) void attention_kernel(
    const __bf16* __restrict__ qkv, __bf16* __restrict__ o)
{
    __shared__ float ks[TSEQ][HDIM];
    __shared__ float vs[TSEQ][HDIM];
    __shared__ float ss[TSEQ][TSEQ];
    int n = blockIdx.x, hh = blockIdx.y, tid = threadIdx.x;
    const float scale = 0.125f;  // 1/sqrt(64)
    for (int i = tid; i < TSEQ * HDIM; i += 256) {
        int t = i / HDIM, d = i % HDIM;
        long base = ((long)(n * TSEQ + t)) * (3 * DMODEL) + hh * HDIM + d;
        ks[t][d] = (float)qkv[base + DMODEL];
        vs[t][d] = (float)qkv[base + 2 * DMODEL];
    }
    __syncthreads();
    for (int i = tid; i < TSEQ * TSEQ; i += 256) {
        int qi = i / TSEQ, ki = i % TSEQ;
        float s;
        if (ki <= qi) {
            const __bf16* qp = &qkv[((long)(n * TSEQ + qi)) * (3 * DMODEL) + hh * HDIM];
            float acc = 0.f;
            #pragma unroll 8
            for (int d = 0; d < HDIM; d++) acc += (float)qp[d] * ks[ki][d];
            s = acc * scale;
        } else s = -1e9f;
        ss[qi][ki] = s;
    }
    __syncthreads();
    if (tid < TSEQ) {
        float mx = -1e30f;
        for (int k = 0; k < TSEQ; k++) mx = fmaxf(mx, ss[tid][k]);
        float sum = 0.f;
        for (int k = 0; k < TSEQ; k++) { float e = expf(ss[tid][k] - mx); ss[tid][k] = e; sum += e; }
        float inv = 1.0f / sum;
        for (int k = 0; k < TSEQ; k++) ss[tid][k] *= inv;
    }
    __syncthreads();
    for (int i = tid; i < TSEQ * HDIM; i += 256) {
        int t = i / HDIM, d = i % HDIM;
        float acc = 0.f;
        for (int k = 0; k < TSEQ; k++) acc += ss[t][k] * vs[k][d];
        o[((long)(n * TSEQ + t)) * DMODEL + hh * HDIM + d] = (__bf16)acc;
    }
}

// ---------------------------------------------------------------------------
// Final: sum 3 components, denormalize, transpose to (B, PRED, M)
// ---------------------------------------------------------------------------
__global__ void final_kernel(
    const float* __restrict__ o2, const float* __restrict__ means,
    const float* __restrict__ stdev, float* __restrict__ out)
{
    int gid = blockIdx.x * 256 + threadIdx.x;
    if (gid >= BATCH * PREDLEN * NVARS) return;
    int m = gid % NVARS;
    int t = (gid / NVARS) % PREDLEN;
    int b = gid / (NVARS * PREDLEN);
    int n = b * NVARS + m;
    long r = (long)n * 3;
    float s = o2[(r + 0) * PREDLEN + t] + o2[(r + 1) * PREDLEN + t] + o2[(r + 2) * PREDLEN + t];
    out[gid] = s * stdev[n] + means[n];
}

// ---------------------------------------------------------------------------
// Host
// ---------------------------------------------------------------------------
static void launch_gemm(const __bf16* A, const __bf16* Bt, const float* bias,
                        const float* res, void* C, int M, int N, int K,
                        int act, int obf, hipStream_t stream)
{
    dim3 grid((N + BN - 1) / BN, (M + BM - 1) / BM);
    gemm_bf16_wmma<<<grid, 256, 0, stream>>>(A, Bt, bias, res, C, M, N, K, act, obf);
}

static void launch_wtrans(const float* W, __bf16* Wt, int K, int N, int Kpad, int Npad,
                          hipStream_t stream)
{
    long total = (long)Npad * Kpad;
    wtrans_kernel<<<(unsigned)((total + 255) / 256), 256, 0, stream>>>(W, Wt, K, N, Kpad, total);
}

extern "C" void kernel_launch(void* const* d_in, const int* in_sizes, int n_in,
                              void* d_out, int out_size, void* d_ws, size_t ws_size,
                              hipStream_t stream)
{
    const float* x       = (const float*)d_in[0];
    const float* in_w    = (const float*)d_in[2];
    const float* in_b    = (const float*)d_in[3];
    const float* pkey    = (const float*)d_in[4];
    const float* wpe     = (const float*)d_in[5];
    const float* ln1_g   = (const float*)d_in[6];
    const float* ln1_b   = (const float*)d_in[7];
    const float* qkv_w   = (const float*)d_in[8];
    const float* qkv_b   = (const float*)d_in[9];
    const float* aproj_w = (const float*)d_in[10];
    const float* aproj_b = (const float*)d_in[11];
    const float* ln2_g   = (const float*)d_in[12];
    const float* ln2_b   = (const float*)d_in[13];
    const float* fc_w    = (const float*)d_in[14];
    const float* fc_b    = (const float*)d_in[15];
    const float* mproj_w = (const float*)d_in[16];
    const float* mproj_b = (const float*)d_in[17];
    const float* lnf_g   = (const float*)d_in[18];
    const float* lnf_b   = (const float*)d_in[19];
    const float* out_w   = (const float*)d_in[20];
    const float* out_b   = (const float*)d_in[21];
    float* out = (float*)d_out;

    float* ws = (float*)d_ws;
    size_t off = 0;
    auto alloc_f32  = [&](size_t n) { float* p = ws + off; off += n; return p; };
    auto alloc_bf16 = [&](size_t n) { __bf16* p = (__bf16*)(ws + off); off += (n + 1) / 2; return p; };

    float*  means  = alloc_f32(NSEQ);
    float*  stdev  = alloc_f32(NSEQ);
    float*  xn     = alloc_f32((size_t)NSEQ * SEQLEN);
    float*  dec    = alloc_f32((size_t)NSEQ * 3 * SEQLEN);
    float*  emb    = alloc_f32((size_t)NSEQ * PNUM * DMODEL);
    float*  keyn   = alloc_f32((size_t)POOL * DMODEL);
    float*  xq     = alloc_f32((size_t)NSEQ * DMODEL);
    int*    idx    = (int*)alloc_f32(NSEQ * TOPK);
    float*  h      = alloc_f32((size_t)NROWS * DMODEL);
    float*  o2     = alloc_f32((size_t)NSEQ * 3 * PREDLEN);

    __bf16* tok_bf = alloc_bf16((size_t)NSEQ * PNUM * 64);
    // tmpA: also viewed as (1344 x 17408) by the output head -> pad to 11*128 rows
    __bf16* tmpA   = alloc_bf16((size_t)1408 * KOUT);       // >= NROWS*DMODEL
    __bf16* wide   = alloc_bf16((size_t)NROWS * FDIM);      // qkv out, fc out
    __bf16* in_wt  = alloc_bf16((size_t)DMODEL * 64);       // (768 x 64)
    __bf16* qkv_wt = alloc_bf16((size_t)NLAYERS * 3 * DMODEL * DMODEL);
    __bf16* apr_wt = alloc_bf16((size_t)NLAYERS * DMODEL * DMODEL);
    __bf16* fc_wt  = alloc_bf16((size_t)NLAYERS * FDIM * DMODEL);
    __bf16* mpr_wt = alloc_bf16((size_t)NLAYERS * DMODEL * FDIM);
    __bf16* out_wt = alloc_bf16((size_t)128 * KOUT);        // N padded 96->128

    dim3 blk(256);

    // 0) weight conversion + transpose (bf16, N-major)
    launch_wtrans(in_w, in_wt, 48, DMODEL, 64, DMODEL, stream);
    launch_wtrans(out_w, out_wt, KOUT, PREDLEN, KOUT, 128, stream);
    for (int l = 0; l < NLAYERS; l++) {
        launch_wtrans(qkv_w  + (long)l * DMODEL * 3 * DMODEL, qkv_wt + (long)l * 3 * DMODEL * DMODEL,
                      DMODEL, 3 * DMODEL, DMODEL, 3 * DMODEL, stream);
        launch_wtrans(aproj_w + (long)l * DMODEL * DMODEL, apr_wt + (long)l * DMODEL * DMODEL,
                      DMODEL, DMODEL, DMODEL, DMODEL, stream);
        launch_wtrans(fc_w   + (long)l * DMODEL * FDIM, fc_wt + (long)l * FDIM * DMODEL,
                      DMODEL, FDIM, DMODEL, FDIM, stream);
        launch_wtrans(mproj_w + (long)l * FDIM * DMODEL, mpr_wt + (long)l * DMODEL * FDIM,
                      FDIM, DMODEL, FDIM, DMODEL, stream);
    }

    // 1) normalize + decompose + patch tokens (bf16, K padded to 64)
    normalize_kernel<<<NSEQ, blk, 0, stream>>>(x, xn, means, stdev);
    decompose_kernel<<<NSEQ, blk, 0, stream>>>(xn, dec);
    {
        long tot = (long)NSEQ * PNUM * 64;
        tok_kernel<<<(unsigned)((tot + 255) / 256), blk, 0, stream>>>(dec, tok_bf);
    }

    // 2) embed: (28672 x 64bf) @ (64 x 768) + in_b -> fp32 emb
    launch_gemm(tok_bf, in_wt, in_b, nullptr, emb, NSEQ * PNUM, DMODEL, 64, 0, 0, stream);

    // 3) prompt retrieval
    l2norm_rows<<<POOL, blk, 0, stream>>>(pkey, keyn, DMODEL);
    xq_kernel<<<NSEQ, blk, 0, stream>>>(emb, xq);
    init_scalar<<<1, blk, 0, stream>>>(out + BATCH * PREDLEN * NVARS);
    topk_kernel<<<NSEQ, blk, 0, stream>>>(xq, keyn, idx, out + BATCH * PREDLEN * NVARS);
    concat_kernel<<<NSEQ, blk, 0, stream>>>(keyn, idx, emb, wpe, h);

    // 4) GPT-2 blocks (bf16 GEMM path, fp32 residual stream)
    for (int l = 0; l < NLAYERS; l++) {
        layernorm_kernel<<<NROWS, blk, 0, stream>>>(h, ln1_g + l * DMODEL, ln1_b + l * DMODEL, tmpA, DMODEL);
        launch_gemm(tmpA, qkv_wt + (long)l * 3 * DMODEL * DMODEL, qkv_b + (long)l * 3 * DMODEL,
                    nullptr, wide, NROWS, 3 * DMODEL, DMODEL, 0, 1, stream);
        attention_kernel<<<dim3(NSEQ, NHEAD), blk, 0, stream>>>(wide, tmpA);
        launch_gemm(tmpA, apr_wt + (long)l * DMODEL * DMODEL, aproj_b + (long)l * DMODEL,
                    h, h, NROWS, DMODEL, DMODEL, 0, 0, stream);
        layernorm_kernel<<<NROWS, blk, 0, stream>>>(h, ln2_g + l * DMODEL, ln2_b + l * DMODEL, tmpA, DMODEL);
        launch_gemm(tmpA, fc_wt + (long)l * FDIM * DMODEL, fc_b + (long)l * FDIM,
                    nullptr, wide, NROWS, FDIM, DMODEL, 1, 1, stream);
        launch_gemm(wide, mpr_wt + (long)l * DMODEL * FDIM, mproj_b + (long)l * DMODEL,
                    h, h, NROWS, DMODEL, FDIM, 0, 0, stream);
    }

    // 5) final LN + output head: (1344 x 17408bf) @ (17408 x 96) -> fp32 o2
    layernorm_kernel<<<NROWS, blk, 0, stream>>>(h, lnf_g, lnf_b, tmpA, DMODEL);
    launch_gemm(tmpA, out_wt, out_b, nullptr, o2, NSEQ * 3, PREDLEN, KOUT, 0, 0, stream);

    // 6) denormalize + transpose
    {
        int tot = BATCH * PREDLEN * NVARS;
        final_kernel<<<(tot + 255) / 256, blk, 0, stream>>>(o2, means, stdev, out);
    }
}
